// SparseDeepGCN_20289425506363
// MI455X (gfx1250) — matrix operand order repaired and checked by reading, not compiled, not run on hardware
//
#include <hip/hip_runtime.h>

typedef __bf16 bf16_t;
typedef __attribute__((ext_vector_type(16))) __bf16 v16bf;
typedef __attribute__((ext_vector_type(8)))  float  v8f;

#define Bc   4
#define Pc   4096
#define NP   16384        // B*P
#define KNN  16
#define Cc   64
#define NBLK 7
#define Fdim 448
#define KD_MAX 96
#define BN_EPS 1e-5f

// ---------------------------------------------------------------------------
// WMMA fragment loaders (wave32; A/B use this file's contiguous k order, 16x16x32 bf16)
// A (16x32): lane l holds row M=l&15; half j -> K = j + (l>>4)*16
// B (32x16): lane l holds col N=l&15; half j -> K = j + (l>>4)*16
// C/D f32  : lane l, vgpr r -> M = r + 8*(l>>4), N = l&15
// ---------------------------------------------------------------------------
__device__ __forceinline__ int a_kmap(int j, int lane) {
    return j + ((lane >> 4) << 4);
}

__device__ __forceinline__ v16bf load_a16(const bf16_t* rowp, int k0, int lane) {
    v16bf a;
#pragma unroll
    for (int j = 0; j < 16; ++j) a[j] = rowp[k0 + a_kmap(j, lane)];
    return a;
}

__device__ __forceinline__ v16bf load_b16(const bf16_t* Bw, int ldb, int k0, int n0,
                                          int N, int lane) {
    int n = n0 + (lane & 15);
    bool ok = (n < N);
    v16bf b;
#pragma unroll
    for (int j = 0; j < 16; ++j) {
        int k = k0 + j + ((lane >> 4) << 4);
        b[j] = ok ? Bw[(size_t)k * ldb + n] : (bf16_t)0.0f;
    }
    return b;
}

// B^T fragment for Gram from an LDS-staged candidate tile [16][64] bf16
__device__ __forceinline__ v16bf lds_bT_gram(const bf16_t* tl, int k0, int lane) {
    const bf16_t* p = tl + (lane & 15) * 64 + k0 + ((lane >> 4) << 4);
    v16bf b;
#pragma unroll
    for (int j = 0; j < 16; ++j) b[j] = p[j];
    return b;
}

__device__ __forceinline__ v8f wmma_bf16(v16bf a, v16bf b, v8f c) {
    return __builtin_amdgcn_wmma_f32_16x16x32_bf16(false, a, false, b, (short)0, c,
                                                   false, false);
}

#define V8F_ZERO {0.f,0.f,0.f,0.f,0.f,0.f,0.f,0.f}

// async global -> LDS staging (gfx1250): GVS addressing, per-lane 16B
__device__ __forceinline__ void async_tile_b128(unsigned lds_off, unsigned voff,
                                                const void* sbase) {
    asm volatile("global_load_async_to_lds_b128 %0, %1, %2"
                 :: "v"(lds_off), "v"(voff), "s"(sbase) : "memory");
}
__device__ __forceinline__ void async_wait_le1() {
    asm volatile("s_wait_asynccnt 1" ::: "memory");
}
__device__ __forceinline__ void async_wait_0() {
    asm volatile("s_wait_asynccnt 0" ::: "memory");
}

// ---------------------------------------------------------------------------
// small utility kernels
// ---------------------------------------------------------------------------
__global__ __launch_bounds__(256) void k_zero(float* p, int n) {
    int i = blockIdx.x * blockDim.x + threadIdx.x;
    if (i < n) p[i] = 0.f;
}

__global__ __launch_bounds__(256) void k_f32_to_bf16(const float* s, bf16_t* d, int n) {
    for (int i = blockIdx.x * blockDim.x + threadIdx.x; i < n; i += gridDim.x * blockDim.x)
        d[i] = (bf16_t)s[i];
}

// pad-convert weights: src f32 [K,N] -> dst bf16 [Kpad,N] (zero pad K rows)
__global__ __launch_bounds__(256) void k_convert_w_pad(const float* src, bf16_t* dst,
                                                       int K, int N, int Kpad) {
    int n = Kpad * N;
    for (int t = blockIdx.x * blockDim.x + threadIdx.x; t < n; t += gridDim.x * blockDim.x) {
        int k = t / N, c = t - k * N;
        dst[t] = (k < K) ? (bf16_t)src[(size_t)k * N + c] : (bf16_t)0.0f;
    }
}

__global__ __launch_bounds__(256) void k_build_x(const float* pos, const float* color,
                                                 float* x, float* pos2) {
    int i = blockIdx.x * blockDim.x + threadIdx.x;
    if (i >= NP) return;
    float px = pos[i * 3 + 0], py = pos[i * 3 + 1], pz = pos[i * 3 + 2];
    x[i * 9 + 0] = px; x[i * 9 + 1] = py; x[i * 9 + 2] = pz;
#pragma unroll
    for (int c = 0; c < 6; ++c) x[i * 9 + 3 + c] = color[i * 6 + c];
    pos2[i] = px * px + py * py + pz * pz;
}

// squared row norms over 64 channels at stride ldx
__global__ __launch_bounds__(256) void k_rownorm(const float* F, int ldx, float* out) {
    int i = blockIdx.x * blockDim.x + threadIdx.x;
    if (i >= NP) return;
    const float* r = F + (size_t)i * ldx;
    float s = 0.f;
#pragma unroll 8
    for (int c = 0; c < Cc; ++c) { float v = r[c]; s += v * v; }
    out[i] = s;
}

// ---------------------------------------------------------------------------
// head KNN on 3-d positions (scalar; trivially cheap next to feature KNN)
// ---------------------------------------------------------------------------
__global__ __launch_bounds__(256) void k_knn_head(const float* pos, const float* pos2,
                                                  int* idx_out) {
    __shared__ float cp[256 * 3];
    int b = blockIdx.x >> 4;                       // 16 blocks per cloud
    int q = ((blockIdx.x & 15) << 8) + threadIdx.x;
    int gq = b * Pc + q;
    float qx = pos[gq * 3 + 0], qy = pos[gq * 3 + 1], qz = pos[gq * 3 + 2];
    float bd[KNN]; int bi[KNN];
#pragma unroll
    for (int s = 0; s < KNN; ++s) { bd[s] = 3.4e38f; bi[s] = 0; }
    float wv = 3.4e38f; int ws = 0;
    for (int t0 = 0; t0 < Pc; t0 += 256) {
        __syncthreads();
        int gj = b * Pc + t0 + threadIdx.x;
        cp[threadIdx.x * 3 + 0] = pos[gj * 3 + 0];
        cp[threadIdx.x * 3 + 1] = pos[gj * 3 + 1];
        cp[threadIdx.x * 3 + 2] = pos[gj * 3 + 2];
        __syncthreads();
        for (int c = 0; c < 256; ++c) {
            int j = t0 + c;
            float dx = qx - cp[c * 3 + 0];
            float dy = qy - cp[c * 3 + 1];
            float dz = qz - cp[c * 3 + 2];
            float dv = dx * dx + dy * dy + dz * dz;
            if (j == q) dv = 1e10f;
            if (dv < wv) {
                bd[ws] = dv; bi[ws] = j;
                wv = -1.f;
#pragma unroll
                for (int s = 0; s < KNN; ++s)
                    if (bd[s] > wv) { wv = bd[s]; ws = s; }
            }
        }
    }
    for (int i = 1; i < KNN; ++i) {
        float dv = bd[i]; int iv = bi[i]; int j = i - 1;
        while (j >= 0 && bd[j] > dv) { bd[j + 1] = bd[j]; bi[j + 1] = bi[j]; --j; }
        bd[j + 1] = dv; bi[j + 1] = iv;
    }
    for (int r = 0; r < KNN; ++r) idx_out[(size_t)gq * KNN + r] = bi[r];
}

// ---------------------------------------------------------------------------
// fused feature-space dilated KNN:
//   candidate tiles DMA'd global->LDS once per block (double-buffered,
//   global_load_async_to_lds_b128 + s_wait_asynccnt), WMMA Gram from LDS,
//   per-query top-k lists kept in the 320KB LDS.
// 128 threads = 4 waves; waves own 4 consecutive query tiles of one cloud
// LDS layout (dynamic): [0,4K) 2 x candidate tile [16][64]bf16
//                       [4K,8K) dist staging [4][16][16]f32
//                       [8K,32K) top dists [4][16][KD_MAX]f32
//                       [+24K) top idx  [4][16][KD_MAX]i32
// ---------------------------------------------------------------------------
#define KNN_SMEM (4096 + 4096 + 4 * 16 * KD_MAX * 4 + 4 * 16 * KD_MAX * 4)

__global__ __launch_bounds__(128) void k_knn_feat(const bf16_t* Fb, const float* x2,
                                                  int* idx_out, int dil) {
    extern __shared__ char smem[];
    bf16_t* tiles  = (bf16_t*)smem;                       // 2 x 2048 B
    float*  stage  = (float*)(smem + 4096);
    float*  td     = (float*)(smem + 8192);
    int*    ti     = (int*)(smem + 8192 + 4 * 16 * KD_MAX * 4);
    int wave = threadIdx.x >> 5;
    int lane = threadIdx.x & 31;
    int tid  = threadIdx.x;
    float* mystage = stage + wave * 256;
    float* mytd = td + wave * 16 * KD_MAX;
    int*   myti = ti + wave * 16 * KD_MAX;

    int qt = blockIdx.x * 4 + wave;            // B*P/16 tiles total
    int b = qt >> 8;                           // 256 query tiles per cloud
    int q0 = (qt & 255) << 4;
    const bf16_t* Fbase = Fb + (size_t)b * Pc * 64;
    const float* x2b = x2 + b * Pc;
    int KD = KNN * dil;

    if (lane < 16)
        for (int s = 0; s < KD; ++s) {
            mytd[lane * KD_MAX + s] = 3.4e38f;
            myti[lane * KD_MAX + s] = 0;
        }
    float wv = 3.4e38f; int ws = 0;

    const bf16_t* arow = Fbase + (size_t)(q0 + (lane & 15)) * 64;
    v16bf a0 = load_a16(arow, 0, lane);
    v16bf a1 = load_a16(arow, 32, lane);

    // prologue: stage candidate tile 0 into buffer 0 (2048B / block, 16B per lane)
    async_tile_b128((unsigned)(tid * 16), (unsigned)(tid * 16), Fbase);

    for (int ct = 0; ct < 256; ++ct) {
        if (ct < 255) {   // stage tile ct+1 into the other buffer, keep 1 in flight
            unsigned voff = (unsigned)(ct + 1) * 2048u + (unsigned)(tid * 16);
            unsigned loff = (unsigned)(((ct + 1) & 1) * 2048) + (unsigned)(tid * 16);
            async_tile_b128(loff, voff, Fbase);
            async_wait_le1();
        } else {
            async_wait_0();
        }
        __syncthreads();                        // tile ct fully in LDS for all waves

        int c0 = ct << 4;
        const bf16_t* tl = tiles + (ct & 1) * 1024;   // elements
        v16bf b0 = lds_bT_gram(tl, 0, lane);
        v16bf b1 = lds_bT_gram(tl, 32, lane);
        v8f acc = V8F_ZERO;
        acc = wmma_bf16(a0, b0, acc);
        acc = wmma_bf16(a1, b1, acc);
#pragma unroll
        for (int r = 0; r < 8; ++r) {
            int qr = r + ((lane >> 4) << 3);
            int cc = c0 + (lane & 15);
            float dv = x2b[q0 + qr] + x2b[cc] - 2.f * acc[r];
            if (q0 + qr == cc) dv = 1e10f;
            mystage[qr * 16 + (lane & 15)] = dv;
        }
        if (lane < 16) {
            for (int c = 0; c < 16; ++c) {
                float dv = mystage[lane * 16 + c];
                if (dv < wv) {
                    mytd[lane * KD_MAX + ws] = dv;
                    myti[lane * KD_MAX + ws] = c0 + c;
                    wv = -1.f;
                    for (int s = 0; s < KD; ++s) {
                        float v = mytd[lane * KD_MAX + s];
                        if (v > wv) { wv = v; ws = s; }
                    }
                }
            }
        }
        __syncthreads();    // all waves done reading buf(ct&1) before it is re-DMA'd
    }

    if (lane < 16) {
        float* L = mytd + lane * KD_MAX;
        int*   I = myti + lane * KD_MAX;
        for (int i = 1; i < KD; ++i) {
            float dv = L[i]; int iv = I[i]; int j = i - 1;
            while (j >= 0 && L[j] > dv) { L[j + 1] = L[j]; I[j + 1] = I[j]; --j; }
            L[j + 1] = dv; I[j + 1] = iv;
        }
        for (int r = 0; r < KNN; ++r)
            idx_out[((size_t)b * Pc + q0 + lane) * KNN + r] = I[r * dil];
    }
}

// ---------------------------------------------------------------------------
// EdgeConv: h = cat(xi, xj-xi) @ W + bias; BN over (B,P,k); relu; max over k
// One wave = one point (16 neighbor rows) x one 16-col tile (nt in 0..3)
// ---------------------------------------------------------------------------
__device__ __forceinline__ v16bf edge_a_tile(const float* xi, const float* xj,
                                             int Cin, int k0, int lane) {
    v16bf a;
#pragma unroll
    for (int j = 0; j < 16; ++j) {
        int kk = k0 + a_kmap(j, lane);
        float v = 0.f;
        if (kk < Cin) v = xi[kk];
        else if (kk < 2 * Cin) v = xj[kk - Cin] - xi[kk - Cin];
        a[j] = (bf16_t)v;
    }
    return a;
}

__device__ __forceinline__ v8f edge_acc(const float* X, int ldx, int Cin,
                                        const int* idxp, const bf16_t* Wb,
                                        int p_global, int b, int nt, int Kpad, int lane) {
    const float* xi = X + (size_t)p_global * ldx;
    int nbr = idxp[(size_t)p_global * KNN + (lane & 15)];
    const float* xj = X + ((size_t)b * Pc + nbr) * (size_t)ldx;
    v8f acc = V8F_ZERO;
    for (int k0 = 0; k0 < Kpad; k0 += 32) {
        v16bf a = edge_a_tile(xi, xj, Cin, k0, lane);
        v16bf bm = load_b16(Wb, Cc, k0, nt * 16, Cc, lane);
        acc = wmma_bf16(a, bm, acc);
    }
    return acc;
}

__global__ __launch_bounds__(256) void k_edge_stats(const float* X, int ldx, int Cin,
                                                    const int* idxp, const bf16_t* Wb,
                                                    const float* bias, float* stats,
                                                    int Kpad) {
    int lane = threadIdx.x & 31;
    int gw = (blockIdx.x * blockDim.x + threadIdx.x) >> 5;
    int nw = (gridDim.x * blockDim.x) >> 5;
    for (int t = gw; t < NP * 4; t += nw) {
        int p_global = t >> 2, nt = t & 3;
        int b = p_global >> 12;
        v8f acc = edge_acc(X, ldx, Cin, idxp, Wb, p_global, b, nt, Kpad, lane);
        int col = nt * 16 + (lane & 15);
        float bv = bias[col];
        float s = 0.f, s2 = 0.f;
#pragma unroll
        for (int r = 0; r < 8; ++r) { float v = acc[r] + bv; s += v; s2 += v * v; }
        s  += __shfl_xor(s, 16, 32);
        s2 += __shfl_xor(s2, 16, 32);
        if (lane < 16) {
            atomicAdd(&stats[col], s);
            atomicAdd(&stats[Cc + col], s2);
        }
    }
}

__global__ __launch_bounds__(256) void k_edge_apply(const float* X, int ldx, int Cin,
                                                    const int* idxp, const bf16_t* Wb,
                                                    const float* bias, const float* stats,
                                                    const float* gamma, const float* beta,
                                                    const float* resid, float* outF,
                                                    bf16_t* outBf, int Kpad) {
    int lane = threadIdx.x & 31;
    int gw = (blockIdx.x * blockDim.x + threadIdx.x) >> 5;
    int nw = (gridDim.x * blockDim.x) >> 5;
    const float inv_cnt = 1.f / (float)(NP * KNN);
    for (int t = gw; t < NP * 4; t += nw) {
        int p_global = t >> 2, nt = t & 3;
        int b = p_global >> 12;
        v8f acc = edge_acc(X, ldx, Cin, idxp, Wb, p_global, b, nt, Kpad, lane);
        int col = nt * 16 + (lane & 15);
        float mu = stats[col] * inv_cnt;
        float var = stats[Cc + col] * inv_cnt - mu * mu;
        float sc = gamma[col] * rsqrtf(var + BN_EPS);
        float be = beta[col];
        float bv = bias[col];
        float m8 = -3.4e38f;
#pragma unroll
        for (int r = 0; r < 8; ++r) {
            float v = (acc[r] + bv - mu) * sc + be;
            v = fmaxf(v, 0.f);
            m8 = fmaxf(m8, v);
        }
        float m = fmaxf(m8, __shfl_xor(m8, 16, 32));   // max over all 16 neighbors
        if (lane < 16) {
            if (resid) m += resid[(size_t)p_global * ldx + col];
            outF[(size_t)p_global * Fdim + col] = m;
            outBf[(size_t)p_global * Cc + col] = (bf16_t)m;
        }
    }
}

// ---------------------------------------------------------------------------
// fusion: max_n relu(feats @ fus_w + fus_b) ; 1 block per 16-row M tile
// ---------------------------------------------------------------------------
__global__ __launch_bounds__(256) void k_fusion(const bf16_t* featsbf, const bf16_t* Wb,
                                                const float* bias, float* fusion) {
    __shared__ float red[8][16];
    int wave = threadIdx.x >> 5;
    int lane = threadIdx.x & 31;
    int m0 = blockIdx.x * 16;
    const bf16_t* rowp = featsbf + (size_t)(m0 + (lane & 15)) * Fdim;
    float vmax[8];
#pragma unroll
    for (int r = 0; r < 8; ++r) vmax[r] = -3.4e38f;
    for (int tt = 0; tt < 8; ++tt) {
        int nt = wave + 8 * tt;            // 64 col tiles of 1024
        v8f acc = V8F_ZERO;
        for (int k0 = 0; k0 < Fdim; k0 += 32) {
            v16bf a = load_a16(rowp, k0, lane);
            v16bf bm = load_b16(Wb, 1024, k0, nt * 16, 1024, lane);
            acc = wmma_bf16(a, bm, acc);
        }
        float bv = bias[nt * 16 + (lane & 15)];
#pragma unroll
        for (int r = 0; r < 8; ++r)
            vmax[r] = fmaxf(vmax[r], fmaxf(acc[r] + bv, 0.f));
    }
#pragma unroll
    for (int r = 0; r < 8; ++r) {
        float v = vmax[r];
        v = fmaxf(v, __shfl_xor(v, 1, 32));
        v = fmaxf(v, __shfl_xor(v, 2, 32));
        v = fmaxf(v, __shfl_xor(v, 4, 32));
        v = fmaxf(v, __shfl_xor(v, 8, 32));
        vmax[r] = v;
    }
    if ((lane & 15) == 0) {
        int g = lane >> 4;
#pragma unroll
        for (int r = 0; r < 8; ++r) red[wave][g * 8 + r] = vmax[r];
    }
    __syncthreads();
    if (threadIdx.x < 16) {
        float m = -3.4e38f;
#pragma unroll
        for (int w = 0; w < 8; ++w) m = fmaxf(m, red[w][threadIdx.x]);
        fusion[m0 + threadIdx.x] = m;
    }
}

// ---------------------------------------------------------------------------
// generic WMMA GEMM: C = relu?(A[M,K]bf16 @ Bw[K,N]bf16 + bias (+ r1v x r1w))
// ---------------------------------------------------------------------------
__global__ __launch_bounds__(256) void k_gemm(const bf16_t* A, int lda, const bf16_t* Bw,
                                              int ldb, const float* bias, const float* r1v,
                                              const float* r1w, float* Cf, bf16_t* Cb,
                                              int M, int N, int Kdim, int relu) {
    int lane = threadIdx.x & 31;
    int gw = (blockIdx.x * blockDim.x + threadIdx.x) >> 5;
    int nw = (gridDim.x * blockDim.x) >> 5;
    int tn_n = (N + 15) >> 4;
    int ntile = ((M + 15) >> 4) * tn_n;
    for (int t = gw; t < ntile; t += nw) {
        int tm = t / tn_n, tn = t - tm * tn_n;
        const bf16_t* rowp = A + (size_t)(tm * 16 + (lane & 15)) * lda;
        v8f acc = V8F_ZERO;
        for (int k0 = 0; k0 < Kdim; k0 += 32) {
            v16bf a = load_a16(rowp, k0, lane);
            v16bf bm = load_b16(Bw, ldb, k0, tn * 16, N, lane);
            acc = wmma_bf16(a, bm, acc);
        }
        int col = tn * 16 + (lane & 15);
#pragma unroll
        for (int r = 0; r < 8; ++r) {
            int rr = tm * 16 + r + ((lane >> 4) << 3);
            if (col < N && rr < M) {
                float v = acc[r] + (bias ? bias[col] : 0.f);
                if (r1v) v += r1v[rr] * r1w[col];
                if (relu) v = fmaxf(v, 0.f);
                if (Cf) Cf[(size_t)rr * N + col] = v;
                if (Cb) Cb[(size_t)rr * N + col] = (bf16_t)v;
            }
        }
    }
}

// ---------------------------------------------------------------------------
// host side
// ---------------------------------------------------------------------------
extern "C" void kernel_launch(void* const* d_in, const int* in_sizes, int n_in,
                              void* d_out, int out_size, void* d_ws, size_t ws_size,
                              hipStream_t stream) {
    const float* pos    = (const float*)d_in[0];
    const float* color  = (const float*)d_in[1];
    const float* head_w = (const float*)d_in[3];
    const float* head_b = (const float*)d_in[4];
    const float* head_g = (const float*)d_in[5];
    const float* head_be= (const float*)d_in[6];
    const float* blk_w  = (const float*)d_in[7];
    const float* blk_b  = (const float*)d_in[8];
    const float* blk_g  = (const float*)d_in[9];
    const float* blk_be = (const float*)d_in[10];
    const float* fus_w  = (const float*)d_in[11];
    const float* fus_b  = (const float*)d_in[12];
    const float* p1_w   = (const float*)d_in[13];
    const float* p1_b   = (const float*)d_in[14];
    const float* p2_w   = (const float*)d_in[15];
    const float* p2_b   = (const float*)d_in[16];
    const float* p3_w   = (const float*)d_in[17];
    const float* p3_b   = (const float*)d_in[18];

    char* base = (char*)d_ws;
    size_t off = 0;
    auto alloc = [&](size_t bytes) -> void* {
        void* p = base + off;
        off += (bytes + 255) & ~(size_t)255;
        return p;
    };
    float*  x       = (float*)alloc((size_t)NP * 9 * 4);
    float*  pos2    = (float*)alloc((size_t)NP * 4);
    float*  x2f     = (float*)alloc((size_t)NP * 4);
    int*    idxb    = (int*)  alloc((size_t)NP * KNN * 4);
    float*  feats   = (float*)alloc((size_t)NP * Fdim * 4);
    bf16_t* fbf     = (bf16_t*)alloc((size_t)NP * Cc * 2);
    bf16_t* featsbf = (bf16_t*)alloc((size_t)NP * Fdim * 2);
    bf16_t* wbuf    = (bf16_t*)alloc((size_t)448 * 1024 * 2);
    float*  stats   = (float*)alloc(2 * Cc * 4);
    float*  fusion  = (float*)alloc((size_t)NP * 4);
    bf16_t* h1bf    = (bf16_t*)alloc((size_t)NP * 512 * 2);
    bf16_t* h2bf    = (bf16_t*)alloc((size_t)NP * 256 * 2);

    // inputs
    k_build_x<<<NP / 256, 256, 0, stream>>>(pos, color, x, pos2);
    k_knn_head<<<NP / 256, 256, 0, stream>>>(pos, pos2, idxb);

    // head EdgeConv (Cin=9, Kpad=32)
    k_convert_w_pad<<<(32 * 64 + 255) / 256, 256, 0, stream>>>(head_w, wbuf, 18, 64, 32);
    k_zero<<<1, 128, 0, stream>>>(stats, 128);
    k_edge_stats<<<512, 256, 0, stream>>>(x, 9, 9, idxb, wbuf, head_b, stats, 32);
    k_edge_apply<<<512, 256, 0, stream>>>(x, 9, 9, idxb, wbuf, head_b, stats, head_g,
                                          head_be, nullptr, feats, fbf, 32);
    k_rownorm<<<NP / 256, 256, 0, stream>>>(feats, Fdim, x2f);

    // residual blocks with dilated dynamic KNN
    for (int i = 0; i < NBLK - 1; ++i) {
        int d = i + 1;
        k_knn_feat<<<256, 128, KNN_SMEM, stream>>>(fbf, x2f, idxb, d);
        k_convert_w_pad<<<(128 * 64 + 255) / 256, 256, 0, stream>>>(
            blk_w + (size_t)i * 128 * 64, wbuf, 128, 64, 128);
        k_zero<<<1, 128, 0, stream>>>(stats, 128);
        k_edge_stats<<<512, 256, 0, stream>>>(feats + i * Cc, Fdim, Cc, idxb, wbuf,
                                              blk_b + i * Cc, stats, 128);
        k_edge_apply<<<512, 256, 0, stream>>>(feats + i * Cc, Fdim, Cc, idxb, wbuf,
                                              blk_b + i * Cc, stats, blk_g + i * Cc,
                                              blk_be + i * Cc, feats + i * Cc,
                                              feats + (i + 1) * Cc, fbf, 128);
        k_rownorm<<<NP / 256, 256, 0, stream>>>(feats + (i + 1) * Cc, Fdim, x2f);
    }

    // fusion + MLP head
    k_f32_to_bf16<<<1024, 256, 0, stream>>>(feats, featsbf, NP * Fdim);
    k_convert_w_pad<<<(448 * 1024 + 255) / 256, 256, 0, stream>>>(fus_w, wbuf, 448, 1024, 448);
    k_fusion<<<NP / 16, 256, 0, stream>>>(featsbf, wbuf, fus_b, fusion);

    k_convert_w_pad<<<(448 * 512 + 255) / 256, 256, 0, stream>>>(p1_w, wbuf, 449, 512, 448);
    k_gemm<<<1024, 256, 0, stream>>>(featsbf, Fdim, wbuf, 512, p1_b, fusion,
                                     p1_w + (size_t)448 * 512, nullptr, h1bf,
                                     NP, 512, 448, 1);
    k_convert_w_pad<<<(512 * 256 + 255) / 256, 256, 0, stream>>>(p2_w, wbuf, 512, 256, 512);
    k_gemm<<<1024, 256, 0, stream>>>(h1bf, 512, wbuf, 256, p2_b, nullptr, nullptr,
                                     nullptr, h2bf, NP, 256, 512, 1);
    k_convert_w_pad<<<(256 * 13 + 255) / 256, 256, 0, stream>>>(p3_w, wbuf, 256, 13, 256);
    k_gemm<<<512, 256, 0, stream>>>(h2bf, 256, wbuf, 13, p3_b, nullptr, nullptr,
                                    (float*)d_out, nullptr, NP, 13, 256, 0);
}
